// GovernanceAwareAttention_63050119905531
// MI455X (gfx1250) — matrix operand order repaired and checked by reading, not compiled, not run on hardware
//
#include <hip/hip_runtime.h>
#include <hip/hip_bf16.h>

typedef __attribute__((ext_vector_type(16))) __bf16 v16bf;
typedef __attribute__((ext_vector_type(8)))  __bf16 v8bf;
typedef __attribute__((ext_vector_type(8)))  float  v8f;

#define DEVINL __device__ __forceinline__

// ---------------------------------------------------------------------------
// WMMA helpers (gfx1250: V_WMMA_F32_16X16X32_BF16, wave32)
// ---------------------------------------------------------------------------
DEVINL v8f wmma_bf16f32(v16bf a, v16bf b, v8f c) {
  return __builtin_amdgcn_wmma_f32_16x16x32_bf16(
      /*neg_a=*/false, a, /*neg_b=*/false, b,
      /*c_mod=*/(short)0, c, /*reuse_a=*/false, /*reuse_b=*/false);
}

// A fragment 16x32: lanes 0-15 = row M, K {kk+0..7, kk+16..23};
//                   lanes 16-31 = row M, K {kk+8..15, kk+24..31}.
DEVINL v16bf frag_a_bf16(const __bf16* __restrict__ row, int kk, int lane) {
  const int c0 = kk + ((lane & 16) ? 8 : 0);
  v8bf lo = *(const v8bf*)(row + c0);
  v8bf hi = *(const v8bf*)(row + c0 + 16);
  v16bf r;
#pragma unroll
  for (int i = 0; i < 8; ++i) { r[i] = lo[i]; r[8 + i] = hi[i]; }
  return r;
}

// B fragment 32x16: lanes 0-15 = column N, K {kk+0..7, kk+16..23};
//                   lanes 16-31 = column N, K {kk+8..15, kk+24..31}.
// Source is a row of B^T (column of B contiguous in memory).
DEVINL v16bf frag_b_bf16(const __bf16* __restrict__ colrow, int kk, int lane) {
  const int c0 = kk + ((lane & 16) ? 8 : 0);
  v8bf lo = *(const v8bf*)(colrow + c0);
  v8bf hi = *(const v8bf*)(colrow + c0 + 16);
  v16bf r;
#pragma unroll
  for (int i = 0; i < 8; ++i) { r[i] = lo[i]; r[8 + i] = hi[i]; }
  return r;
}

// ---------------------------------------------------------------------------
// Small prep kernels
// ---------------------------------------------------------------------------
// tw[b,h] = trust[b,:] @ Wt[:,h] + bt[h]   (B*NH = 32 values)
__global__ void k_trust(const float* __restrict__ trust, const float* __restrict__ Wt,
                        const float* __restrict__ bt, float* __restrict__ tw) {
  int t = threadIdx.x;
  if (t >= 32) return;
  int b = t >> 4, h = t & 15;
  float s = bt[h];
#pragma unroll 4
  for (int i = 0; i < 64; ++i) s += trust[b * 64 + i] * Wt[i * 16 + h];
  tw[t] = s;
}

// y[i] = (bf16)x[i]
__global__ void k_cvt(const float* __restrict__ x, __bf16* __restrict__ y, int n) {
  int idx = blockIdx.x * 256 + threadIdx.x;
  if (idx < n) y[idx] = (__bf16)x[idx];
}

// Wt_out[n*K + k] = (bf16)W[k*N + n]   (weight transpose + f32->bf16)
__global__ void k_wtrans(const float* __restrict__ W, __bf16* __restrict__ Wt,
                         int K, int N) {
  int idx = blockIdx.x * 256 + threadIdx.x;
  if (idx >= K * N) return;
  int k = idx / N, n = idx - k * N;
  Wt[(size_t)n * K + k] = (__bf16)W[idx];
}

// Vt[b,h,d,s] = V[b,s,h,d]   (bf16, makes columns of V contiguous for WMMA B)
__global__ void k_vtrans(const __bf16* __restrict__ V, __bf16* __restrict__ Vt) {
  int idx = blockIdx.x * 256 + threadIdx.x;       // over 2*1024*16*64 = 2^21
  if (idx >= (2 << 20)) return;
  int d = idx & 63, h = (idx >> 6) & 15, s = (idx >> 10) & 1023, b = idx >> 20;
  Vt[((((size_t)(b * 16 + h)) * 64 + d) << 10) + s] = V[idx];
}

// ---------------------------------------------------------------------------
// GEMM: out[M,N] = A_bf16[M,K] @ B (+ bias),  B supplied as bf16 B^T rows.
// 256 threads = 8 waves; block tile 128x128; wave tile 64x32 (4x2 wmma tiles).
// 2-deep software pipeline over k (ping-pong, K % 64 == 0).
// ---------------------------------------------------------------------------
template <bool OUT_BF16>
__global__ __launch_bounds__(256) void k_gemm_ab(
    const __bf16* __restrict__ A, const __bf16* __restrict__ Bt,
    const float* __restrict__ bias, void* __restrict__ out, int N, int K) {
  const int lane = threadIdx.x & 31, wid = threadIdx.x >> 5;
  const int m0 = blockIdx.y * 128 + (wid >> 2) * 64;
  const int n0 = blockIdx.x * 128 + (wid & 3) * 32;

  v8f acc[4][2] = {};
  const __bf16* arow[4];
  const __bf16* brow[2];
#pragma unroll
  for (int mt = 0; mt < 4; ++mt)
    arow[mt] = A + (size_t)(m0 + mt * 16 + (lane & 15)) * K;
#pragma unroll
  for (int nt = 0; nt < 2; ++nt)
    brow[nt] = Bt + (size_t)(n0 + nt * 16 + (lane & 15)) * K;

  v16bf a0[4], b0[2], a1[4], b1[2];
#pragma unroll
  for (int mt = 0; mt < 4; ++mt) a0[mt] = frag_a_bf16(arow[mt], 0, lane);
#pragma unroll
  for (int nt = 0; nt < 2; ++nt) b0[nt] = frag_b_bf16(brow[nt], 0, lane);

  for (int kk = 0; kk < K; kk += 64) {
    // issue loads for kk+32 while computing on kk
#pragma unroll
    for (int mt = 0; mt < 4; ++mt) a1[mt] = frag_a_bf16(arow[mt], kk + 32, lane);
#pragma unroll
    for (int nt = 0; nt < 2; ++nt) b1[nt] = frag_b_bf16(brow[nt], kk + 32, lane);
#pragma unroll
    for (int mt = 0; mt < 4; ++mt)
#pragma unroll
      for (int nt = 0; nt < 2; ++nt)
        acc[mt][nt] = wmma_bf16f32(a0[mt], b0[nt], acc[mt][nt]);
    // issue loads for kk+64 (wrapped on last iter) while computing on kk+32
    const int kn = (kk + 64 < K) ? (kk + 64) : 0;
#pragma unroll
    for (int mt = 0; mt < 4; ++mt) a0[mt] = frag_a_bf16(arow[mt], kn, lane);
#pragma unroll
    for (int nt = 0; nt < 2; ++nt) b0[nt] = frag_b_bf16(brow[nt], kn, lane);
#pragma unroll
    for (int mt = 0; mt < 4; ++mt)
#pragma unroll
      for (int nt = 0; nt < 2; ++nt)
        acc[mt][nt] = wmma_bf16f32(a1[mt], b1[nt], acc[mt][nt]);
  }

#pragma unroll
  for (int mt = 0; mt < 4; ++mt)
#pragma unroll
    for (int nt = 0; nt < 2; ++nt)
#pragma unroll
      for (int r = 0; r < 8; ++r) {
        int m = m0 + mt * 16 + r + ((lane & 16) ? 8 : 0);
        int n = n0 + nt * 16 + (lane & 15);
        float v = acc[mt][nt][r] + bias[n];
        if (OUT_BF16) ((__bf16*)out)[(size_t)m * N + n] = (__bf16)v;
        else          ((float*)out)[(size_t)m * N + n] = v;
      }
}

// ---------------------------------------------------------------------------
// CS[b,q,k] = alpha * CQ[b,q,:2048] . CK[b,k,:2048]
// (mean over constitutional heads folded into one K=2048 GEMM, alpha=scale/CH)
// ---------------------------------------------------------------------------
__global__ __launch_bounds__(256) void k_gemm_cs(
    const __bf16* __restrict__ CQ, const __bf16* __restrict__ CK,
    float* __restrict__ CS, float alpha) {
  const int lane = threadIdx.x & 31, wid = threadIdx.x >> 5;
  const int b = blockIdx.z;
  const int m0 = blockIdx.y * 128 + (wid >> 2) * 64;
  const int n0 = blockIdx.x * 128 + (wid & 3) * 32;
  const int K = 2048;

  v8f acc[4][2] = {};
  const __bf16* arow[4];
  const __bf16* brow[2];
#pragma unroll
  for (int mt = 0; mt < 4; ++mt)
    arow[mt] = CQ + (size_t)(b * 1024 + m0 + mt * 16 + (lane & 15)) * K;
#pragma unroll
  for (int nt = 0; nt < 2; ++nt)
    brow[nt] = CK + (size_t)(b * 1024 + n0 + nt * 16 + (lane & 15)) * K;

  v16bf a0[4], b0[2], a1[4], b1[2];
#pragma unroll
  for (int mt = 0; mt < 4; ++mt) a0[mt] = frag_a_bf16(arow[mt], 0, lane);
#pragma unroll
  for (int nt = 0; nt < 2; ++nt) b0[nt] = frag_b_bf16(brow[nt], 0, lane);

  for (int kk = 0; kk < K; kk += 64) {
#pragma unroll
    for (int mt = 0; mt < 4; ++mt) a1[mt] = frag_a_bf16(arow[mt], kk + 32, lane);
#pragma unroll
    for (int nt = 0; nt < 2; ++nt) b1[nt] = frag_b_bf16(brow[nt], kk + 32, lane);
#pragma unroll
    for (int mt = 0; mt < 4; ++mt)
#pragma unroll
      for (int nt = 0; nt < 2; ++nt)
        acc[mt][nt] = wmma_bf16f32(a0[mt], b0[nt], acc[mt][nt]);
    const int kn = (kk + 64 < K) ? (kk + 64) : 0;
#pragma unroll
    for (int mt = 0; mt < 4; ++mt) a0[mt] = frag_a_bf16(arow[mt], kn, lane);
#pragma unroll
    for (int nt = 0; nt < 2; ++nt) b0[nt] = frag_b_bf16(brow[nt], kn, lane);
#pragma unroll
    for (int mt = 0; mt < 4; ++mt)
#pragma unroll
      for (int nt = 0; nt < 2; ++nt)
        acc[mt][nt] = wmma_bf16f32(a1[mt], b1[nt], acc[mt][nt]);
  }

#pragma unroll
  for (int mt = 0; mt < 4; ++mt)
#pragma unroll
    for (int nt = 0; nt < 2; ++nt)
#pragma unroll
      for (int r = 0; r < 8; ++r) {
        int m = m0 + mt * 16 + r + ((lane & 16) ? 8 : 0);
        int n = n0 + nt * 16 + (lane & 15);
        CS[((size_t)b * 1024 + m) * 1024 + n] = alpha * acc[mt][nt][r];
      }
}

// ---------------------------------------------------------------------------
// Fused attention: per (b, h, 32-row q block)
//   scores = tw*(scale*Q@K^T + mask) + CS ; softmax ; probs out ; ctx = P@V
// LDS: probs tile f32 (128 KB) + bf16 mirror (64 KB) = 192 KB of 320 KB WGP LDS.
// ---------------------------------------------------------------------------
__global__ __launch_bounds__(256) void k_attn(
    const __bf16* __restrict__ Q, const __bf16* __restrict__ Kb,
    const __bf16* __restrict__ Vt, const float* __restrict__ mask,
    const float* __restrict__ CS, const float* __restrict__ tw,
    float* __restrict__ probs, __bf16* __restrict__ ctx) {
  extern __shared__ char smem[];
  float*  ptile = (float*)smem;                       // [32][1024] f32
  __bf16* pbf   = (__bf16*)(smem + 32 * 1024 * 4);    // [32][1024] bf16
  const int lane = threadIdx.x & 31, wid = threadIdx.x >> 5;
  const int b = blockIdx.z, h = blockIdx.y, q0 = blockIdx.x * 32;
  const float scale = 0.125f;  // 1/sqrt(64)
  const float twv = tw[b * 16 + h];

  // ---- Phase 1: scores (32 x 1024), K = HD = 64 (2 k-steps) ----
  {
    v16bf aq[2][2];
#pragma unroll
    for (int mt = 0; mt < 2; ++mt) {
      const __bf16* qrow =
          Q + (((size_t)(b * 1024 + q0 + mt * 16 + (lane & 15))) * 16 + h) * 64;
#pragma unroll
      for (int ks = 0; ks < 2; ++ks) aq[mt][ks] = frag_a_bf16(qrow, ks * 32, lane);
    }
    const int ncol0 = wid * 128;  // each wave owns 128 score columns
#pragma unroll
    for (int nt = 0; nt < 8; ++nt) {
      const int nn = ncol0 + nt * 16;
      const __bf16* krow =
          Kb + (((size_t)(b * 1024 + nn + (lane & 15))) * 16 + h) * 64;
      v8f acc[2] = {};
#pragma unroll
      for (int ks = 0; ks < 2; ++ks) {
        v16bf bk = frag_b_bf16(krow, ks * 32, lane);
#pragma unroll
        for (int mt = 0; mt < 2; ++mt)
          acc[mt] = wmma_bf16f32(aq[mt][ks], bk, acc[mt]);
      }
#pragma unroll
      for (int mt = 0; mt < 2; ++mt)
#pragma unroll
        for (int r = 0; r < 8; ++r) {
          int lq = mt * 16 + r + ((lane & 16) ? 8 : 0);
          int n = nn + (lane & 15);
          int qg = q0 + lq;
          size_t off = ((size_t)b * 1024 + qg) * 1024 + n;
          ptile[lq * 1024 + n] =
              twv * (scale * acc[mt][r] + mask[off]) + CS[off];
        }
    }
  }
  __syncthreads();

  // ---- Phase 2: row softmax over 1024, write probs (f32) + bf16 LDS copy ----
  {
#pragma unroll
    for (int rr = 0; rr < 4; ++rr) {
      int lq = wid * 4 + rr;
      float mx = -3.0e38f;
      for (int j = lane; j < 1024; j += 32) mx = fmaxf(mx, ptile[lq * 1024 + j]);
#pragma unroll
      for (int o = 16; o >= 1; o >>= 1) mx = fmaxf(mx, __shfl_xor(mx, o, 32));
      float sum = 0.f;
      for (int j = lane; j < 1024; j += 32) {
        float e = __expf(ptile[lq * 1024 + j] - mx);
        ptile[lq * 1024 + j] = e;
        sum += e;
      }
#pragma unroll
      for (int o = 16; o >= 1; o >>= 1) sum += __shfl_xor(sum, o, 32);
      float inv = 1.0f / sum;
      float* prow = probs + (((size_t)(b * 16 + h)) * 1024 + (q0 + lq)) * 1024;
      for (int j = lane; j < 1024; j += 32) {
        float pv = ptile[lq * 1024 + j] * inv;
        pbf[lq * 1024 + j] = (__bf16)pv;
        prow[j] = pv;
      }
    }
  }
  __syncthreads();

  // ---- Phase 3: ctx(32x64) = P(32x1024) @ V(1024x64), K = 1024 ----
  {
    const int mt = wid >> 2;        // 0..1 : 16-row tile
    const int d0 = (wid & 3) * 16;  // 0..48 : 16-col tile of head_dim
    const __bf16* arow = &pbf[(mt * 16 + (lane & 15)) * 1024];
    const __bf16* vrow =
        Vt + ((((size_t)(b * 16 + h)) * 64 + d0 + (lane & 15)) << 10);
    v8f acc = {};
    v16bf a0 = frag_a_bf16(arow, 0, lane);
    v16bf v0 = frag_b_bf16(vrow, 0, lane);
    for (int kk = 0; kk < 1024; kk += 64) {
      v16bf a1 = frag_a_bf16(arow, kk + 32, lane);
      v16bf v1 = frag_b_bf16(vrow, kk + 32, lane);
      acc = wmma_bf16f32(a0, v0, acc);
      const int kn = (kk + 64 < 1024) ? (kk + 64) : 0;
      a0 = frag_a_bf16(arow, kn, lane);
      v0 = frag_b_bf16(vrow, kn, lane);
      acc = wmma_bf16f32(a1, v1, acc);
    }
#pragma unroll
    for (int r = 0; r < 8; ++r) {
      int lq = mt * 16 + r + ((lane & 16) ? 8 : 0);
      int d = d0 + (lane & 15);
      ctx[((size_t)(b * 1024 + q0 + lq)) * 1024 + h * 64 + d] = (__bf16)acc[r];
    }
  }
}

// ---------------------------------------------------------------------------
// Host-side orchestration
// ---------------------------------------------------------------------------
extern "C" void kernel_launch(void* const* d_in, const int* in_sizes, int n_in,
                              void* d_out, int out_size, void* d_ws,
                              size_t ws_size, hipStream_t stream) {
  const float* x     = (const float*)d_in[0];   // [2,1024,1024]
  const float* mask  = (const float*)d_in[1];   // [2,1,1024,1024]
  const float* trust = (const float*)d_in[2];   // [2,64]
  const float* Wq  = (const float*)d_in[3];  const float* bq  = (const float*)d_in[4];
  const float* Wk  = (const float*)d_in[5];  const float* bk  = (const float*)d_in[6];
  const float* Wv  = (const float*)d_in[7];  const float* bv  = (const float*)d_in[8];
  const float* Wcq = (const float*)d_in[9];  const float* bcq = (const float*)d_in[10];
  const float* Wck = (const float*)d_in[11]; const float* bck = (const float*)d_in[12];
  // d_in[13]/d_in[14] = Wcv/bcv: dead in the reference (cv is never used)
  const float* Wt  = (const float*)d_in[15]; const float* bt  = (const float*)d_in[16];
  const float* Wo  = (const float*)d_in[17]; const float* bo  = (const float*)d_in[18];
  (void)in_sizes; (void)n_in; (void)out_size; (void)ws_size;

  float* out   = (float*)d_out;                       // [2,1024,1024]
  float* probs = out + (size_t)2 * 1024 * 1024;       // [2,16,1024,1024]

  char* p = (char*)d_ws;
  auto alloc = [&](size_t bytes) -> char* {
    char* r = p;
    p += (bytes + 255) & ~(size_t)255;
    return r;
  };
  const size_t M = 2048;  // B*S
  __bf16* xbf  = (__bf16*)alloc(M * 1024 * 2);
  __bf16* WqT  = (__bf16*)alloc(1024 * 1024 * 2);
  __bf16* WkT  = (__bf16*)alloc(1024 * 1024 * 2);
  __bf16* WvT  = (__bf16*)alloc(1024 * 1024 * 2);
  __bf16* WoT  = (__bf16*)alloc(1024 * 1024 * 2);
  __bf16* WcqT = (__bf16*)alloc(2048 * 1024 * 2);
  __bf16* WckT = (__bf16*)alloc(2048 * 1024 * 2);
  __bf16* Qbf  = (__bf16*)alloc(M * 1024 * 2);
  __bf16* Kbf  = (__bf16*)alloc(M * 1024 * 2);
  __bf16* Vbf  = (__bf16*)alloc(M * 1024 * 2);
  __bf16* Vtbf = (__bf16*)alloc(M * 1024 * 2);
  __bf16* CQbf = (__bf16*)alloc(M * 2048 * 2);
  __bf16* CKbf = (__bf16*)alloc(M * 2048 * 2);
  __bf16* ctxb = (__bf16*)alloc(M * 1024 * 2);
  float*  CSb  = (float*)alloc((size_t)2 * 1024 * 1024 * 4);
  float*  twb  = (float*)alloc(32 * 4);

  // 1) input conversion, weight transposes (f32 -> bf16^T), trust GEMV
  int g1 = (1024 * 1024 + 255) / 256, g2 = (2048 * 1024 + 255) / 256;
  k_cvt<<<g2, 256, 0, stream>>>(x, xbf, 2048 * 1024);
  k_wtrans<<<g1, 256, 0, stream>>>(Wq, WqT, 1024, 1024);
  k_wtrans<<<g1, 256, 0, stream>>>(Wk, WkT, 1024, 1024);
  k_wtrans<<<g1, 256, 0, stream>>>(Wv, WvT, 1024, 1024);
  k_wtrans<<<g1, 256, 0, stream>>>(Wo, WoT, 1024, 1024);
  k_wtrans<<<g2, 256, 0, stream>>>(Wcq, WcqT, 1024, 2048);
  k_wtrans<<<g2, 256, 0, stream>>>(Wck, WckT, 1024, 2048);
  k_trust<<<1, 32, 0, stream>>>(trust, Wt, bt, twb);

  // 2) projections (pure-bf16 WMMA loops, bf16 outputs)
  k_gemm_ab<true><<<dim3(8, 16), 256, 0, stream>>>(xbf, WqT, bq, Qbf, 1024, 1024);
  k_gemm_ab<true><<<dim3(8, 16), 256, 0, stream>>>(xbf, WkT, bk, Kbf, 1024, 1024);
  k_gemm_ab<true><<<dim3(8, 16), 256, 0, stream>>>(xbf, WvT, bv, Vbf, 1024, 1024);
  k_gemm_ab<true><<<dim3(16, 16), 256, 0, stream>>>(xbf, WcqT, bcq, CQbf, 2048, 1024);
  k_gemm_ab<true><<<dim3(16, 16), 256, 0, stream>>>(xbf, WckT, bck, CKbf, 2048, 1024);

  // 3) V transpose for the P@V WMMA B operand
  k_vtrans<<<(2 << 20) / 256, 256, 0, stream>>>(Vbf, Vtbf);

  // 4) mean constitutional scores as one K=2048 GEMM (alpha = scale/CH)
  k_gemm_cs<<<dim3(8, 8, 2), 256, 0, stream>>>(CQbf, CKbf, CSb, 0.125f / 32.0f);

  // 5) fused score + softmax + probs + P@V  (192 KB dynamic LDS)
  k_attn<<<dim3(32, 16, 2), 256, 32 * 1024 * (4 + 2), stream>>>(
      Qbf, Kbf, Vtbf, mask, CSb, twb, probs, ctxb);

  // 6) output projection (f32 out + bias)
  k_gemm_ab<false><<<dim3(8, 16), 256, 0, stream>>>(ctxb, WoT, bo, out, 1024, 1024);
}
